// AttentionSequencePoolingLayer_88648124989839
// MI455X (gfx1250) — matrix operand order repaired and checked by reading, not compiled, not run on hardware
//
#include <hip/hip_runtime.h>
#include <hip/hip_bf16.h>

typedef __attribute__((ext_vector_type(16))) _Float16     v16h;
typedef __attribute__((ext_vector_type(8)))  float        v8f;
typedef __attribute__((ext_vector_type(4)))  unsigned int u32x4;
typedef __attribute__((ext_vector_type(8)))  int          i32x8;
typedef __attribute__((ext_vector_type(4)))  int          i32x4;

#define B_   2048
#define S_   200
#define D_   64
#define H1_  64
#define H2_  16
#define EPS_ 1e-9f
#define NT_  13                      // ceil(S/16) sequence tiles

// ---- dynamic-LDS layout (base offset 0 -> compile-time TDM lds addresses) ----
#define KSROW_     (D_ + 4)                       // padded row: 68 floats
#define KSBUF_B_   (16 * KSROW_ * 4)              // one 16x68 f32 buffer: 4352 B
#define KS_OFF_    0
#define QS_OFF_    (KS_OFF_ + 2 * KSBUF_B_)       // 8704
#define H1ROW_     (H1_ + 8)                      // padded row: 72 halves
#define H1_OFF_    (QS_OFF_ + D_ * 4)             // 8960
#define SC_OFF_    (H1_OFF_ + 16 * H1ROW_ * 2)    // 11264
#define SMEM_B_    (SC_OFF_ + 16 * 4)             // 11328 bytes total

#if __has_builtin(__builtin_amdgcn_tensor_load_to_lds)
#define USE_TDM 1
#else
#define USE_TDM 0
#endif

// ---------------------------------------------------------------------------
// One-time weight folding + conversion f32 -> f16 into workspace.
//   att@W1 = q@(W1a+W1c) + k@(W1b-W1c) + (q*k)@W1d
// ---------------------------------------------------------------------------
__global__ void cvt_weights_kernel(const float* __restrict__ W1,
                                   const float* __restrict__ W2,
                                   _Float16* __restrict__ W1q,
                                   _Float16* __restrict__ W1k,
                                   _Float16* __restrict__ W1p,
                                   _Float16* __restrict__ W2h) {
    const int i = blockIdx.x * blockDim.x + threadIdx.x;
    if (i < D_ * H1_) {
        const float a = W1[0 * D_ * H1_ + i];
        const float b = W1[1 * D_ * H1_ + i];
        const float c = W1[2 * D_ * H1_ + i];
        const float d = W1[3 * D_ * H1_ + i];
        W1q[i] = (_Float16)(a + c);
        W1k[i] = (_Float16)(b - c);
        W1p[i] = (_Float16)d;
    }
    if (i < H1_ * H2_) W2h[i] = (_Float16)W2[i];
}

__device__ __forceinline__ float dice_f(float x, float al, float mu, float rv) {
    float xn = (x - mu) * rv;
    float p  = 1.0f / (1.0f + __expf(-xn));
    return x * (p + al * (1.0f - p));
}

#if USE_TDM
// ---------------------------------------------------------------------------
// TDM: DMA one 16x64 f32 key tile global->LDS. Padding 4 DWORDs every 64
// DWORDs reproduces the [16][68] bank-conflict-free layout; rows beyond
// tensor_dim1 read as zero (tail-tile masking for free). ISA cdna5 §8.
// This toolchain's builtin takes 6 args: g0, g1, g2, g3, <extra v8i>, cpol.
// ---------------------------------------------------------------------------
__device__ __forceinline__ void tdm_issue_tile(const float* gsrc,
                                               unsigned lds_byte_addr,
                                               int rows_remaining) {
    const unsigned long long ga = (unsigned long long)(uintptr_t)gsrc;
    u32x4 g0;
    g0[0] = 1u;                                            // count=1, user mode
    g0[1] = lds_byte_addr;                                 // LDS dest (bytes)
    g0[2] = (unsigned)ga;                                  // global_addr[31:0]
    g0[3] = (unsigned)((ga >> 32) & 0x01FFFFFFull)         // global_addr[56:32]
          | (2u << 30);                                    // type=2 ("image")
    i32x8 g1;
    g1[0] = (int)((2u << 16)      // data_size = 4 bytes
                | (1u << 20)      // pad_enable
                | (5u << 22)      // pad_interval: 64 DWORDs
                | (3u << 25));    // pad_amount : 4 DWORDs
    g1[1] = (int)(((unsigned)D_) << 16);                   // tensor_dim0 = 64
    g1[2] = (int)(((unsigned)rows_remaining & 0xFFFFu) << 16); // tensor_dim1
    g1[3] = (int)(((unsigned)D_) << 16);                   // tile_dim0 = 64
    g1[4] = 16;                                            // tile_dim1=16, tile_dim2=0
    g1[5] = D_;                                            // tensor_dim0_stride = 64
    g1[6] = 0;
    g1[7] = 0;
    i32x4 g2 = {1, 1, 0, 0};                 // tensor_dim2=1, tensor_dim3=1
    i32x4 g3 = {0, (int)(1u << 16), 0, 0};   // tensor_dim4=1
    i32x8 gz = {0, 0, 0, 0, 0, 0, 0, 0};     // unused trailing group
    __builtin_amdgcn_tensor_load_to_lds(g0, g1, g2, g3, gz, 0);
}
#endif

// ---------------------------------------------------------------------------
// Main fused kernel: one block per batch row, 4 waves (128 threads, wave32)
// ---------------------------------------------------------------------------
__global__ __launch_bounds__(128)
void din_attn_kernel(const float* __restrict__ q_emb,
                     const float* __restrict__ k_emb,
                     const int*   __restrict__ seq_len,
                     const _Float16* __restrict__ W1q,
                     const _Float16* __restrict__ W1k,
                     const _Float16* __restrict__ W1p,
                     const float* __restrict__ alpha1,
                     const float* __restrict__ mean1,
                     const float* __restrict__ var1,
                     const _Float16* __restrict__ W2h,
                     const float* __restrict__ alpha2,
                     const float* __restrict__ mean2,
                     const float* __restrict__ var2,
                     const float* __restrict__ W3,
                     float* __restrict__ out) {
    extern __shared__ char smem[];
    float    (*ks)[16][KSROW_] = (float (*)[16][KSROW_])(smem + KS_OFF_);
    float*    qs               = (float*)(smem + QS_OFF_);
    _Float16 (*h1s)[H1ROW_]    = (_Float16 (*)[H1ROW_])(smem + H1_OFF_);
    float*    scores           = (float*)(smem + SC_OFF_);

    const int b    = blockIdx.x;
    const int tid  = threadIdx.x;
    const int lane = tid & 31;
    const int wv   = tid >> 5;   // wave id 0..3, owns H1 columns [16*wv, 16*wv+16)
    const int lm   = lane & 15;  // M index (A/C rows) or N index (B/C cols)
    const int lh   = lane >> 4;  // half-wave selector for K sub-ranges
    const int seqlen = seq_len[b];
    const float* kbase = k_emb + (size_t)b * S_ * D_;

#if USE_TDM
    // prologue DMA: tile 0 -> buffer 0 (overlaps weight-fragment setup below)
    if (wv == 0) tdm_issue_tile(kbase, KS_OFF_, S_);
#endif

    if (tid < D_) qs[tid] = q_emb[b * D_ + tid];

    // ---- hoist folded-W1 B-fragments: K dim is 64 per matrix (2 frags each) ----
    v16h w1qb[2], w1kb[2], w1pb[2];
#pragma unroll
    for (int kk = 0; kk < 2; ++kk) {
#pragma unroll
        for (int j = 0; j < 16; ++j) {
            const int K   = 32 * kk + ((j < 8) ? (8 * lh + j) : (16 + 8 * lh + (j - 8)));          // 0..63
            const int idx = K * H1_ + 16 * wv + lm;
            w1qb[kk][j] = W1q[idx];
            w1kb[kk][j] = W1k[idx];
            w1pb[kk][j] = W1p[idx];
        }
    }
    // ---- W2 B-fragments ----
    v16h w2b[2];
#pragma unroll
    for (int kk = 0; kk < 2; ++kk)
#pragma unroll
        for (int j = 0; j < 16; ++j) {
            const int K = 32 * kk + ((j < 8) ? (8 * lh + j) : (16 + 8 * lh + (j - 8)));            // 0..63
            w2b[kk][j] = W2h[K * H2_ + lm];
        }

    // Dice params are per-output-column == per-lane scalars in the C/D layout
    const int   n1  = 16 * wv + lm;
    const float al1 = alpha1[n1];
    const float mu1 = mean1[n1];
    const float rv1 = rsqrtf(var1[n1] + EPS_);
    const float al2 = alpha2[lm];
    const float mu2 = mean2[lm];
    const float rv2 = rsqrtf(var2[lm] + EPS_);
    const float w3v = W3[lm];

    __syncthreads();   // qs visible

    // ---- per-lane q values in registers (A-fragment K order), and the
    //      sequence-invariant q @ (W1a+W1c) partial accumulator ----
    float qf[2][16];
    v8f accq = {0.f, 0.f, 0.f, 0.f, 0.f, 0.f, 0.f, 0.f};
#pragma unroll
    for (int kk = 0; kk < 2; ++kk) {
        v16h aq;
#pragma unroll
        for (int j = 0; j < 16; ++j) {
            const int jj = (j < 8) ? (8 * lh + j) : (16 + 8 * lh + (j - 8));
            const float qv = qs[32 * kk + jj];
            qf[kk][j] = qv;
            aq[j] = (_Float16)qv;
        }
        accq = __builtin_amdgcn_wmma_f32_16x16x32_f16(false, aq, false, w1qb[kk],
                                                      (short)0, accq, false, false);
    }

    float oacc = 0.0f;   // thread tid<64 owns output dim d=tid

    for (int t = 0; t < NT_; ++t) {
        const int base = t * 16;
        const int buf  = t & 1;

#if USE_TDM
        // publish buffer `buf`: issuing wave drains TENSORcnt, then barrier
        // makes the DMA'd tile (and all prev-iteration LDS reads) visible.
        __builtin_amdgcn_s_wait_tensorcnt(0);
        __syncthreads();
        // pipeline: kick off DMA of tile t+1 into the other buffer; it runs
        // on the TDM engine while we do WMMA/Dice on this one.
        if (t + 1 < NT_ && wv == 0)
            tdm_issue_tile(kbase + (size_t)(base + 16) * D_,
                           (unsigned)(KS_OFF_ + (buf ^ 1) * KSBUF_B_),
                           S_ - (base + 16));
#else
        __syncthreads();   // previous-iteration readers done
#pragma unroll
        for (int r = 0; r < 2; ++r) {
            const int idx = tid + 128 * r;        // float4 index 0..255
            const int row = idx >> 4, c4 = idx & 15;
            const int s = base + row;
            float4 v = make_float4(0.f, 0.f, 0.f, 0.f);
            if (s < S_) v = ((const float4*)kbase)[(size_t)s * (D_ / 4) + c4];
            *(float4*)&ks[buf][row][c4 * 4] = v;
        }
        __syncthreads();
#endif

        // ---- GEMM1: acc = q-part + k @ W1k' + (q*k) @ W1d ----
        v8f acc = accq;
#pragma unroll
        for (int kk = 0; kk < 2; ++kk) {
            v16h ak, ap;
#pragma unroll
            for (int j = 0; j < 16; ++j) {
                const int jj = (j < 8) ? (8 * lh + j) : (16 + 8 * lh + (j - 8));
                const float kv = ks[buf][lm][32 * kk + jj];
                ak[j] = (_Float16)kv;
                ap[j] = (_Float16)(qf[kk][j] * kv);
            }
            acc = __builtin_amdgcn_wmma_f32_16x16x32_f16(false, ak, false, w1kb[kk],
                                                         (short)0, acc, false, false);
            acc = __builtin_amdgcn_wmma_f32_16x16x32_f16(false, ap, false, w1pb[kk],
                                                         (short)0, acc, false, false);
        }

        // dice1 + stage h1 tile as f16 (C layout: row = i + 8*lh, col = n1)
#pragma unroll
        for (int i = 0; i < 8; ++i) {
            const float h = dice_f(acc[i], al1, mu1, rv1);
            h1s[i + 8 * lh][n1] = (_Float16)h;
        }
        __syncthreads();

        // ---- GEMM2 + dice2 + score, wave 0 only (uniform branch: EXEC all-1) ----
        if (wv == 0) {
            v8f acc2 = {0.f, 0.f, 0.f, 0.f, 0.f, 0.f, 0.f, 0.f};
#pragma unroll
            for (int kk = 0; kk < 2; ++kk) {
                v16h a2;
#pragma unroll
                for (int j = 0; j < 16; ++j) {
                    const int jj = (j < 8) ? (8 * lh + j) : (16 + 8 * lh + (j - 8));
                    a2[j] = h1s[lm][32 * kk + jj];
                }
                acc2 = __builtin_amdgcn_wmma_f32_16x16x32_f16(false, a2, false, w2b[kk],
                                                              (short)0, acc2, false, false);
            }
#pragma unroll
            for (int i = 0; i < 8; ++i) {
                const float h2   = dice_f(acc2[i], al2, mu2, rv2);
                float       part = h2 * w3v;
#pragma unroll
                for (int off = 1; off < 16; off <<= 1)
                    part += __shfl_xor(part, off, 16);
                if (lm == 0) {
                    const int m = i + 8 * lh;
                    const int s = base + m;
                    scores[m] = (s < seqlen) ? (1.0f / (1.0f + __expf(-part))) : 0.0f;
                }
            }
        }
        __syncthreads();

        // ---- weighted accumulation: out[b,:] += sum_r score[r]*key[r,:] ----
        if (tid < D_) {
#pragma unroll
            for (int r = 0; r < 16; ++r)
                oacc += scores[r] * ks[buf][r][tid];
        }
    }

    if (tid < D_) out[b * D_ + tid] = oacc;
}

// ---------------------------------------------------------------------------
extern "C" void kernel_launch(void* const* d_in, const int* in_sizes, int n_in,
                              void* d_out, int out_size, void* d_ws, size_t ws_size,
                              hipStream_t stream) {
    const float* q    = (const float*)d_in[0];
    const float* kemb = (const float*)d_in[1];
    const int*   sl   = (const int*)d_in[2];
    const float* W1   = (const float*)d_in[3];
    const float* a1   = (const float*)d_in[4];
    const float* m1   = (const float*)d_in[5];
    const float* v1   = (const float*)d_in[6];
    const float* W2   = (const float*)d_in[7];
    const float* a2   = (const float*)d_in[8];
    const float* m2   = (const float*)d_in[9];
    const float* v2   = (const float*)d_in[10];
    const float* W3   = (const float*)d_in[11];
    float* out = (float*)d_out;

    _Float16* W1q = (_Float16*)d_ws;          // 64*64 halves = 8 KB
    _Float16* W1k = W1q + D_ * H1_;           // 8 KB
    _Float16* W1p = W1k + D_ * H1_;           // 8 KB
    _Float16* W2h = W1p + D_ * H1_;           // 2 KB

    cvt_weights_kernel<<<16, 256, 0, stream>>>(W1, W2, W1q, W1k, W1p, W2h);
    din_attn_kernel<<<B_, 128, SMEM_B_, stream>>>(q, kemb, sl, W1q, W1k, W1p,
                                                  a1, m1, v1, W2h, a2, m2, v2,
                                                  W3, out);
}
